// GaussianDiffusion_85315230368164
// MI455X (gfx1250) — matrix operand order repaired and checked by reading, not compiled, not run on hardware
//
#include <hip/hip_runtime.h>
#include <math.h>

// ---------------------------------------------------------------------------
// PatchCore scoring for MI455X (gfx1250, wave32, WMMA).
//
// Heavy op: D[N,M] = ||e_i - m_j|| via  xn_i - 2*(E.M^T)_ij + yn_j, with the
// N x M x 384 GEMM done with v_wmma_f32_16x16x32_f16 and the row-min/argmin
// fused into the WMMA epilogue.  K-outer / row-tile-inner loop keeps live
// VGPRs ~130 (no spills, no vgpr-msb churn); an opaque asm dependency pins
// the A-fragment ds_loads inside the column loop instead of hoist-and-spill.
// ---------------------------------------------------------------------------

typedef __attribute__((ext_vector_type(16))) _Float16 v16h;
typedef __attribute__((ext_vector_type(8)))  float    v8f;

#define DIMS            384
#define KSTEPS          12          // 384 / 32
#define ROWS_PER_BLOCK  64
#define ROW_TILES       4           // 64 / 16
#define WAVES           8
#define BLOCK_THREADS   256
#define BATCH           8
#define KNN             9
#define BIGF            3.0e38f

union HFrag {
    v16h     v;
    _Float16 h[16];
    float4   q[2];     // two 16-byte chunks (8 halves each)
};

// ---------------------------------------------------------------------------
// Row L2-norms: one wave per row.
// ---------------------------------------------------------------------------
__global__ __launch_bounds__(BLOCK_THREADS)
void row_norms_kernel(const float* __restrict__ x, float* __restrict__ out, int rows)
{
    int wave = blockIdx.x * (BLOCK_THREADS / 32) + (threadIdx.x >> 5);
    int lane = threadIdx.x & 31;
    if (wave >= rows) return;
    const float* r = x + (size_t)wave * DIMS;
    float s = 0.0f;
    for (int c = lane; c < DIMS; c += 32) { float v = r[c]; s = fmaf(v, v, s); }
    #pragma unroll
    for (int m = 16; m >= 1; m >>= 1) s += __shfl_xor(s, m, 32);
    if (lane == 0) out[wave] = s;
}

// ---------------------------------------------------------------------------
// Main fused distance + row-min/argmin kernel.
// Block: 256 threads (8 waves), owns 64 embedding rows (A tile in LDS, f16).
// Each wave walks a strided 1/8 slice of the 2048 column tiles (16 mem rows
// each).  Per K-step: one B fragment built from global fp32 (reused across
// the 4 row tiles), 4 A fragments streamed from LDS, 4 WMMAs into 4 live
// accumulators.  Min key in the loop is (yn_j - 2*S_ij); xn and sqrt are
// applied once per row at the end (monotone per row).
// ---------------------------------------------------------------------------
__global__ __launch_bounds__(BLOCK_THREADS)
void dist_min_kernel(const float* __restrict__ emb, const float* __restrict__ mem,
                     const float* __restrict__ xn,  const float* __restrict__ yn,
                     float* __restrict__ ps_out, int* __restrict__ loc_out,
                     int N, int M)
{
    __shared__ _Float16 As[ROWS_PER_BLOCK * DIMS];          // 48 KB
    __shared__ float    redval[WAVES][ROWS_PER_BLOCK];      //  2 KB
    __shared__ int      redidx[WAVES][ROWS_PER_BLOCK];      //  2 KB

    const int tid  = threadIdx.x;
    const int wave = tid >> 5;
    const int lane = tid & 31;
    const int half = lane >> 4;      // which 16-lane half
    const int l16  = lane & 15;
    const int row0 = blockIdx.x * ROWS_PER_BLOCK;

    // Stage the 64 x 384 A tile into LDS as f16.
    for (int i = tid; i < ROWS_PER_BLOCK * DIMS; i += BLOCK_THREADS) {
        int r = i / DIMS, c = i - r * DIMS;
        float v = (row0 + r < N) ? emb[(size_t)(row0 + r) * DIMS + c] : 0.0f;
        As[i] = (_Float16)v;
    }
    __syncthreads();

    float best[ROW_TILES][8];
    int   bidx[ROW_TILES][8];
    #pragma unroll
    for (int rt = 0; rt < ROW_TILES; ++rt)
        #pragma unroll
        for (int i = 0; i < 8; ++i) { best[rt][i] = BIGF; bidx[rt][i] = 0x7fffffff; }

    const int kk = 8 * half;                 // ISA A layout: K half-split by lane half
    const int numColTiles = M >> 4;

    for (int ct = wave; ct < numColTiles; ct += WAVES) {
        const int    myCol = (ct << 4) + l16;               // B lane = column (mem row)
        const float* brow  = mem + (size_t)myCol * DIMS;
        const float  ynv   = yn[myCol];

        // Opaque zero: compiler cannot prove the LDS addresses are invariant
        // across ct, so the A-fragment ds_loads stay inside this loop.
        int aoff = 0;
        asm volatile("" : "+v"(aoff));

        v8f c[ROW_TILES];
        #pragma unroll
        for (int rt = 0; rt < ROW_TILES; ++rt)
            c[rt] = (v8f){0.f, 0.f, 0.f, 0.f, 0.f, 0.f, 0.f, 0.f};

        #pragma unroll
        for (int ks = 0; ks < KSTEPS; ++ks) {
            // B fragment (same layout as A: elements 0-7 hold K 8*half..+7,
            // elements 8-15 hold K 16+8*half..+7).
            HFrag b;
            const int kb = ks * 32 + 8 * half;
            #pragma unroll
            for (int v = 0; v < 16; ++v)
                b.h[v] = (_Float16)brow[kb + (v < 8 ? v : v + 8)];

            #pragma unroll
            for (int rt = 0; rt < ROW_TILES; ++rt) {
                HFrag a;
                const _Float16* ar =
                    As + (size_t)(rt * 16 + l16) * DIMS + ks * 32 + kk + aoff;
                a.q[0] = *reinterpret_cast<const float4*>(ar);        // K kk..kk+7
                a.q[1] = *reinterpret_cast<const float4*>(ar + 16);   // K 16+kk..
                c[rt] = __builtin_amdgcn_wmma_f32_16x16x32_f16(
                            false, a.v, false, b.v, (short)0, c[rt], false, false);
            }
        }

        // c[rt][i] = dot(row rt*16 + i + 8*half, col myCol).
        // Min key: yn - 2*dot  (per-row monotone in the true distance).
        #pragma unroll
        for (int rt = 0; rt < ROW_TILES; ++rt)
            #pragma unroll
            for (int i = 0; i < 8; ++i) {
                float t = fmaf(-2.0f, c[rt][i], ynv);
                if (t < best[rt][i]) { best[rt][i] = t; bidx[rt][i] = myCol; }
            }
    }

    // Reduce the 16 columns within each lane-half, then across waves via LDS.
    #pragma unroll
    for (int rt = 0; rt < ROW_TILES; ++rt)
        #pragma unroll
        for (int i = 0; i < 8; ++i) {
            float v = best[rt][i]; int ix = bidx[rt][i];
            #pragma unroll
            for (int m = 8; m >= 1; m >>= 1) {   // masks < 16 stay within the half
                float ov = __shfl_xor(v, m, 32);
                int   oi = __shfl_xor(ix, m, 32);
                if (ov < v || (ov == v && oi < ix)) { v = ov; ix = oi; }
            }
            if (l16 == 0) {
                int r = rt * 16 + i + 8 * half;
                redval[wave][r] = v; redidx[wave][r] = ix;
            }
        }
    __syncthreads();

    if (tid < ROWS_PER_BLOCK) {
        float v = redval[0][tid]; int ix = redidx[0][tid];
        #pragma unroll
        for (int w = 1; w < WAVES; ++w) {
            float ov = redval[w][tid]; int oi = redidx[w][tid];
            if (ov < v || (ov == v && oi < ix)) { v = ov; ix = oi; }
        }
        int r = row0 + tid;
        if (r < N) {
            ps_out[r] = sqrtf(fmaxf(xn[r] + v, 0.0f));   // restore xn + sqrt
            loc_out[r] = ix;
        }
    }
}

// ---------------------------------------------------------------------------
// Per-batch argmax of patch scores; record score, patch index, NN mem index.
// ---------------------------------------------------------------------------
__global__ __launch_bounds__(BLOCK_THREADS)
void batch_select_kernel(const float* __restrict__ ps, const int* __restrict__ loc,
                         float* __restrict__ score_out, int* __restrict__ maxp_out,
                         int* __restrict__ nnidx_out, int P)
{
    const int b = blockIdx.x;
    const float* p = ps + (size_t)b * P;
    __shared__ float sv[BLOCK_THREADS];
    __shared__ int   si[BLOCK_THREADS];

    float v = -BIGF; int ix = 0x7fffffff;
    for (int i = threadIdx.x; i < P; i += BLOCK_THREADS) {
        float x = p[i];
        if (x > v || (x == v && i < ix)) { v = x; ix = i; }
    }
    sv[threadIdx.x] = v; si[threadIdx.x] = ix;
    __syncthreads();
    for (int s = BLOCK_THREADS / 2; s >= 1; s >>= 1) {
        if (threadIdx.x < s) {
            float ov = sv[threadIdx.x + s]; int oi = si[threadIdx.x + s];
            if (ov > sv[threadIdx.x] || (ov == sv[threadIdx.x] && oi < si[threadIdx.x])) {
                sv[threadIdx.x] = ov; si[threadIdx.x] = oi;
            }
        }
        __syncthreads();
    }
    if (threadIdx.x == 0) {
        score_out[b] = sv[0];
        maxp_out[b]  = si[0];
        nnidx_out[b] = loc[(size_t)b * P + si[0]];
    }
}

// ---------------------------------------------------------------------------
// d2[b][m] = || mem[nn[b]] - mem[m] ||   (tiny: 8 x 32768 x 384)
// ---------------------------------------------------------------------------
__global__ __launch_bounds__(BLOCK_THREADS)
void d2_kernel(const float* __restrict__ mem, const int* __restrict__ nnidx,
               const float* __restrict__ yn, float* __restrict__ d2, int M)
{
    const int b = blockIdx.y;
    __shared__ float q[DIMS];
    __shared__ float qn;
    const int nn = nnidx[b];
    for (int i = threadIdx.x; i < DIMS; i += BLOCK_THREADS)
        q[i] = mem[(size_t)nn * DIMS + i];
    if (threadIdx.x == 0) qn = yn[nn];
    __syncthreads();

    const int m = blockIdx.x * BLOCK_THREADS + threadIdx.x;
    if (m < M) {
        const float* r = mem + (size_t)m * DIMS;
        float dot = 0.0f;
        #pragma unroll 4
        for (int c = 0; c < DIMS; ++c) dot = fmaf(q[c], r[c], dot);
        float dd = qn - 2.0f * dot + yn[m];
        d2[(size_t)b * M + m] = sqrtf(fmaxf(dd, 0.0f));
    }
}

// ---------------------------------------------------------------------------
// Per batch: 9 sequential argmin passes over d2 (top-k ascending), d3
// distances from the max-feature to the support vectors, softmax, score.
// ---------------------------------------------------------------------------
__global__ __launch_bounds__(BLOCK_THREADS)
void topk_score_kernel(float* __restrict__ d2, const float* __restrict__ mem,
                       const float* __restrict__ emb, const int* __restrict__ maxp,
                       const float* __restrict__ score, float* __restrict__ pred_out,
                       int M, int P)
{
    const int b = blockIdx.x;
    float* db = d2 + (size_t)b * M;
    __shared__ float sv[BLOCK_THREADS];
    __shared__ int   si[BLOCK_THREADS];
    __shared__ int   support[KNN];
    __shared__ float d3[KNN];

    for (int k = 0; k < KNN; ++k) {
        float v = BIGF; int ix = 0x7fffffff;
        for (int i = threadIdx.x; i < M; i += BLOCK_THREADS) {
            float x = db[i];
            if (x < v || (x == v && i < ix)) { v = x; ix = i; }
        }
        sv[threadIdx.x] = v; si[threadIdx.x] = ix;
        __syncthreads();
        for (int s = BLOCK_THREADS / 2; s >= 1; s >>= 1) {
            if (threadIdx.x < s) {
                float ov = sv[threadIdx.x + s]; int oi = si[threadIdx.x + s];
                if (ov < sv[threadIdx.x] || (ov == sv[threadIdx.x] && oi < si[threadIdx.x])) {
                    sv[threadIdx.x] = ov; si[threadIdx.x] = oi;
                }
            }
            __syncthreads();
        }
        if (threadIdx.x == 0) { support[k] = si[0]; db[si[0]] = BIGF; }
        __syncthreads();
    }

    const int wave = threadIdx.x >> 5, lane = threadIdx.x & 31;
    const float* mf = emb + ((size_t)b * P + maxp[b]) * DIMS;
    for (int k = wave; k < KNN; k += WAVES) {
        const float* sr = mem + (size_t)support[k] * DIMS;
        float s = 0.0f;
        for (int c = lane; c < DIMS; c += 32) { float d = mf[c] - sr[c]; s = fmaf(d, d, s); }
        #pragma unroll
        for (int m = 16; m >= 1; m >>= 1) s += __shfl_xor(s, m, 32);
        if (lane == 0) d3[k] = sqrtf(fmaxf(s, 0.0f));
    }
    __syncthreads();

    if (threadIdx.x == 0) {
        float mx = d3[0];
        for (int k = 1; k < KNN; ++k) mx = fmaxf(mx, d3[k]);
        float sum = 0.0f;
        for (int k = 0; k < KNN; ++k) sum += expf(d3[k] - mx);
        float p0 = expf(d3[0] - mx) / sum;
        pred_out[b] = (1.0f - p0) * score[b];
    }
}

// ---------------------------------------------------------------------------
extern "C" void kernel_launch(void* const* d_in, const int* in_sizes, int n_in,
                              void* d_out, int out_size, void* d_ws, size_t ws_size,
                              hipStream_t stream)
{
    const float* emb = (const float*)d_in[0];
    const float* mem = (const float*)d_in[1];
    // Scalars live on device; shapes are fixed by setup_inputs (D=384, B=8).
    const int N = in_sizes[0] / DIMS;     // 6272
    const int M = in_sizes[1] / DIMS;     // 32768
    const int B = BATCH;                  // 8
    const int P = N / B;                  // 784

    // Workspace layout (all 16B-aligned for these sizes).
    char* w = (char*)d_ws;
    float* xn    = (float*)w;  w += (size_t)N * sizeof(float);
    float* yn    = (float*)w;  w += (size_t)M * sizeof(float);
    int*   loc   = (int*)w;    w += (size_t)N * sizeof(int);
    float* d2    = (float*)w;  w += (size_t)B * M * sizeof(float);
    float* score = (float*)w;  w += 16 * sizeof(float);
    int*   maxp  = (int*)w;    w += 16 * sizeof(int);
    int*   nnidx = (int*)w;    w += 16 * sizeof(int);
    (void)ws_size; (void)n_in; (void)out_size;

    float* ps_out   = (float*)d_out;        // patch_map, flat [N]
    float* pred_out = (float*)d_out + N;    // pred_score [B]

    const int wavesPerBlock = BLOCK_THREADS / 32;

    row_norms_kernel<<<(N + wavesPerBlock - 1) / wavesPerBlock, BLOCK_THREADS, 0, stream>>>(emb, xn, N);
    row_norms_kernel<<<(M + wavesPerBlock - 1) / wavesPerBlock, BLOCK_THREADS, 0, stream>>>(mem, yn, M);

    dist_min_kernel<<<(N + ROWS_PER_BLOCK - 1) / ROWS_PER_BLOCK, BLOCK_THREADS, 0, stream>>>(
        emb, mem, xn, yn, ps_out, loc, N, M);

    batch_select_kernel<<<B, BLOCK_THREADS, 0, stream>>>(ps_out, loc, score, maxp, nnidx, P);

    dim3 g2((M + BLOCK_THREADS - 1) / BLOCK_THREADS, B);
    d2_kernel<<<g2, BLOCK_THREADS, 0, stream>>>(mem, nnidx, yn, d2, M);

    topk_score_kernel<<<B, BLOCK_THREADS, 0, stream>>>(d2, mem, emb, maxp, score, pred_out, M, P);
}
